// Lstm_60748017434907
// MI455X (gfx1250) — matrix operand order repaired and checked by reading, not compiled, not run on hardware
//
#include <hip/hip_runtime.h>

// ---------------------------------------------------------------------------
// MI455X (gfx1250) LSTM: 2-layer recurrence over 64 steps, batch 256, H=1024.
// bf16 WMMA (v_wmma_f32_16x16x32_bf16) everywhere.
//  - Input GEMMs (16384x4096x1024) precomputed device-wide; A staged in LDS as
//    16x128 slabs via GLOBAL_LOAD_ASYNC_TO_LDS_B128 (ASYNCcnt path) when the
//    toolchain exposes it, else synchronous b128 copy.
//  - XG kept in WMMA-fragment-tiled layout ([tile][lane][8] bf16) so producer
//    and consumer touch it only with b128 vectors; marked non-temporal so the
//    128MB stream does not evict the L2-resident weights (192MB L2).
//  - Recurrence: h in double-buffered LDS, c in VGPRs, Whh streamed from L2.
// ---------------------------------------------------------------------------

typedef __bf16 bf16;
typedef __bf16 v16bf __attribute__((ext_vector_type(16)));
typedef __bf16 v8bf  __attribute__((ext_vector_type(8)));
typedef __bf16 v4bf  __attribute__((ext_vector_type(4)));
typedef float  v8f   __attribute__((ext_vector_type(8)));

#define LSTM_S 64      // sequence length (original B)
#define LSTM_N 256     // recurrence batch (original T)
#define LSTM_D 1024
#define LSTM_H 1024
#define LSTM_G 4096    // 4*H gate columns
#define NCLS   27
#define MROWS  (LSTM_S * LSTM_N)   // 16384 rows for the input GEMMs
#define NT_G   (LSTM_G / 16)       // 256 column tiles in XG

#ifndef __has_builtin
#define __has_builtin(x) 0
#endif

#if __has_builtin(__builtin_amdgcn_global_load_async_to_lds_b128)
#define HAVE_ASYNC_LDS 1
typedef int v4i_gcc __attribute__((vector_size(16)));   // matches builtin proto
static __device__ __forceinline__ void async_copy_b128(void* lds, const void* g) {
  // global -> LDS DMA, tracked by ASYNCcnt (CDNA5 ISA)
  __builtin_amdgcn_global_load_async_to_lds_b128((v4i_gcc*)g, (v4i_gcc*)lds, 0, 0);
}
#if __has_builtin(__builtin_amdgcn_s_wait_asynccnt)
#define WAIT_ASYNC() __builtin_amdgcn_s_wait_asynccnt(0)
#else
#define WAIT_ASYNC() asm volatile("s_wait_asynccnt 0x0" ::: "memory")
#endif
#endif

static __device__ __forceinline__ v16bf cat8(v8bf lo, v8bf hi) {
  return __builtin_shufflevector(lo, hi, 0,1,2,3,4,5,6,7,8,9,10,11,12,13,14,15);
}
static __device__ __forceinline__ v8f wmma_bf16(v16bf a, v16bf b, v8f c) {
  // (neg_a, A, neg_b, B, c_mod, C, reuse_a, reuse_b)
  return __builtin_amdgcn_wmma_f32_16x16x32_bf16(false, a, false, b, (short)0, c,
                                                 false, false);
}
static __device__ __forceinline__ float sigm(float x) {
  return 1.f / (1.f + __expf(-x));
}
static __device__ __forceinline__ float tanh_f(float x) {
  x = fminf(fmaxf(x, -15.f), 15.f);
  float e = __expf(2.f * x);
  return (e - 1.f) / (e + 1.f);
}

// ---------------------------------------------------------------------------
// fp32 -> bf16 conversion, 4 elements/thread (all sizes are multiples of 4)
// ---------------------------------------------------------------------------
__global__ void cvt_f32_bf16(const float4* __restrict__ s, v4bf* __restrict__ d,
                             int n4) {
  int i = blockIdx.x * blockDim.x + threadIdx.x;
  int stride = gridDim.x * blockDim.x;
  for (; i < n4; i += stride) {
    float4 v = s[i];
    v4bf o;
    o[0] = (bf16)v.x; o[1] = (bf16)v.y; o[2] = (bf16)v.z; o[3] = (bf16)v.w;
    d[i] = o;
  }
}

// ---------------------------------------------------------------------------
// XG(tiled) = A[M,1024](bf16,row-major) @ W[4096,1024].T + bih + bhh
// grid (G/512, M/16), block 256 (8 waves); each wave: 16x64 tile.
// A staged in LDS as 16x128 K-slabs: one async b128 per thread per slab.
// XG stored in WMMA D-fragment tile layout: ((mt*NT_G + nt)*32 + lane)*8.
// ---------------------------------------------------------------------------
__global__ __launch_bounds__(256) void gemm_xg(const bf16* __restrict__ A,
                                               const bf16* __restrict__ W,
                                               const float* __restrict__ bih,
                                               const float* __restrict__ bhh,
                                               bf16* __restrict__ XG) {
  __shared__ __align__(16) bf16 atile[16][128];   // 4KB K-slab
  const int lane  = threadIdx.x & 31;
  const int wv    = threadIdx.x >> 5;          // 0..7
  const int m0    = blockIdx.y * 16;
  const int nbase = blockIdx.x * 512 + wv * 64;
  // ISA fragment layouts (CDNA5 ISA §7.12.2):
  const int mA  = lane & 15;                   // A: lane -> M row
  const int kbA = (lane >> 4) * 8;             // A: K slabs [kb,kb+8) and [16+kb,..)
  const int nB  = lane & 15;                   // B: lane -> N col
  const int kbB = (lane >> 4) * 8;             // B: K slabs [kb,kb+8) and [16+kb,..)

  v8f acc[4] = {};

  const int arow = threadIdx.x >> 4;           // 0..15
  const int acol = (threadIdx.x & 15) * 8;     // 0..120

  for (int k0 = 0; k0 < LSTM_D; k0 += 128) {
    __syncthreads();   // previous slab fully consumed
#ifdef HAVE_ASYNC_LDS
    async_copy_b128(&atile[arow][acol],
                    &A[(size_t)(m0 + arow) * LSTM_D + k0 + acol]);
    WAIT_ASYNC();      // my lane's LDS write landed; barrier covers the rest
#else
    *(v8bf*)&atile[arow][acol] =
        *(const v8bf*)&A[(size_t)(m0 + arow) * LSTM_D + k0 + acol];
#endif
    __syncthreads();   // slab visible to all waves
    if (k0 + 128 < LSTM_D) {
      // pull next weight slab toward the WGP while we crunch this one
      __builtin_prefetch(W + (size_t)(nbase + nB) * LSTM_D + k0 + 128 + kbB, 0, 3);
    }
#pragma unroll
    for (int kk = 0; kk < 128; kk += 32) {
      v16bf af = cat8(*(const v8bf*)&atile[mA][kk + kbA],
                      *(const v8bf*)&atile[mA][kk + 16 + kbA]);
#pragma unroll
      for (int t = 0; t < 4; ++t) {
        const bf16* wp =
            W + (size_t)(nbase + t * 16 + nB) * LSTM_D + k0 + kk + kbB;
        v16bf bf_ = cat8(*(const v8bf*)wp, *(const v8bf*)(wp + 16));
        acc[t] = wmma_bf16(af, bf_, acc[t]);
      }
    }
  }
  // epilogue: bias + bf16, one non-temporal b128 store per tile
  const int mt = blockIdx.y;
#pragma unroll
  for (int t = 0; t < 4; ++t) {
    const int n0 = nbase + t * 16;
    const int nt = n0 >> 4;
    float bias = bih[n0 + nB] + bhh[n0 + nB];
    v8bf ov;
#pragma unroll
    for (int g = 0; g < 8; ++g) ov[g] = (bf16)(acc[t][g] + bias);
    v8bf* dst = (v8bf*)(XG + ((size_t)(mt * NT_G + nt) * 32 + lane) * 8);
    __builtin_nontemporal_store(ov, dst);      // streaming: keep L2 for weights
  }
}

// ---------------------------------------------------------------------------
// Recurrence: 64 steps. One WG per 16 batch rows; 512 threads = 16 waves;
// each wave owns 64 hidden columns (4 tiles), computes all 4 gates for them.
// h: LDS double buffer (2*16*1024 bf16 = 64KB of the 320KB WGP LDS).
// c: persistent accumulator VGPRs. Whh (8MB bf16) streamed from L2 each step.
// XG read back in fragment-tiled layout: one b128 (non-temporal) per gate.
// ---------------------------------------------------------------------------
__global__ __launch_bounds__(512) void lstm_rec(const bf16* __restrict__ XG,
                                                const bf16* __restrict__ Whh,
                                                bf16* __restrict__ Hout) {
  __shared__ __align__(16) bf16 hbuf[2][16][LSTM_H];
  const int lane = threadIdx.x & 31;
  const int wv   = threadIdx.x >> 5;           // 0..15
  const int bt   = blockIdx.x;                 // batch tile (16 rows)
  const int t0   = bt * 16;
  const int mA  = lane & 15;
  const int kbA = (lane >> 4) * 8;
  const int nB  = lane & 15;
  const int kbB = (lane >> 4) * 8;
  const int mD  = (lane >> 4) * 8;

  for (int i = threadIdx.x; i < 16 * LSTM_H; i += blockDim.x)
    (&hbuf[0][0][0])[i] = (bf16)0.f;           // h0 = 0
  __syncthreads();

  v8f cst[4] = {};                             // c0 = 0 (per-wave exclusive cols)

  for (int s = 0; s < LSTM_S; ++s) {
    const int cur = s & 1, nxt = cur ^ 1;
    const int mt = s * 16 + bt;                // XG row tile for this step
#pragma unroll
    for (int tt = 0; tt < 4; ++tt) {
      const int nb = wv * 64 + tt * 16;        // hidden column base
      v8f acc[4];
      // init accumulators from fragment-tiled XG (includes bih+bhh)
#pragma unroll
      for (int q = 0; q < 4; ++q) {
        const int nt = q * 64 + (nb >> 4);
        const v8bf* xp =
            (const v8bf*)(XG + ((size_t)(mt * NT_G + nt) * 32 + lane) * 8);
        v8bf xv = __builtin_nontemporal_load(xp);
        v8f a0;
#pragma unroll
        for (int g = 0; g < 8; ++g) a0[g] = (float)xv[g];
        acc[q] = a0;
      }
      // gates += h @ Whh.T   (K = 1024, 32 WMMA k-steps, A frag shared by gates)
      for (int k0 = 0; k0 < LSTM_H; k0 += 32) {
        v16bf af = cat8(*(const v8bf*)&hbuf[cur][mA][k0 + kbA],
                        *(const v8bf*)&hbuf[cur][mA][k0 + 16 + kbA]);
#pragma unroll
        for (int q = 0; q < 4; ++q) {
          const bf16* wp =
              Whh + (size_t)(q * LSTM_H + nb + nB) * LSTM_H + k0 + kbB;
          v16bf bf_ = cat8(*(const v8bf*)wp, *(const v8bf*)(wp + 16));
          acc[q] = wmma_bf16(af, bf_, acc[q]);
        }
      }
      // pointwise LSTM cell update (gate order i, f, g, o)
#pragma unroll
      for (int g = 0; g < 8; ++g) {
        float iv = acc[0][g], fv = acc[1][g], gv = acc[2][g], ov = acc[3][g];
        float c = sigm(fv) * cst[tt][g] + sigm(iv) * tanh_f(gv);
        float h = sigm(ov) * tanh_f(c);
        cst[tt][g] = c;
        const int m = mD + g;
        const int col = nb + nB;
        bf16 hb = (bf16)h;
        hbuf[nxt][m][col] = hb;                // transpose-ish: scalar b16
        Hout[(size_t)(s * LSTM_N + t0 + m) * LSTM_H + col] = hb;
      }
    }
    __syncthreads();                           // publish h for next step
  }
}

// ---------------------------------------------------------------------------
// FC head: out[b,:] = relu(H2row(b) @ W1.T + b1) @ W2.T + b2,  b = 0..63.
// H2 row for output b is r = b*256 + 255. Tiny (70 MFLOP) -> fp32 kernel.
// ---------------------------------------------------------------------------
__global__ __launch_bounds__(512) void fc_head(const bf16* __restrict__ H2,
                                               const float* __restrict__ W1,
                                               const float* __restrict__ b1,
                                               const float* __restrict__ W2,
                                               const float* __restrict__ b2,
                                               float* __restrict__ out) {
  __shared__ float f1[LSTM_H / 2];
  const int b = blockIdx.x;
  const int j = threadIdx.x;                   // 512 threads = H/2
  const bf16* hrow = H2 + ((size_t)b * LSTM_N + (LSTM_N - 1)) * LSTM_H;
  float s = b1[j];
  const float* w = W1 + (size_t)j * LSTM_H;
  for (int k = 0; k < LSTM_H; k += 4) {
    s += (float)hrow[k] * w[k] + (float)hrow[k + 1] * w[k + 1] +
         (float)hrow[k + 2] * w[k + 2] + (float)hrow[k + 3] * w[k + 3];
  }
  f1[j] = fmaxf(s, 0.f);
  __syncthreads();
  if (j < NCLS) {
    float s2 = b2[j];
    const float* w2 = W2 + (size_t)j * (LSTM_H / 2);
    for (int k = 0; k < LSTM_H / 2; ++k) s2 += f1[k] * w2[k];
    out[b * NCLS + j] = s2;
  }
}

// ---------------------------------------------------------------------------
extern "C" void kernel_launch(void* const* d_in, const int* in_sizes, int n_in,
                              void* d_out, int out_size, void* d_ws,
                              size_t ws_size, hipStream_t stream) {
  const float* x    = (const float*)d_in[0];
  const float* Wih0 = (const float*)d_in[1];
  const float* Whh0 = (const float*)d_in[2];
  const float* bih0 = (const float*)d_in[3];
  const float* bhh0 = (const float*)d_in[4];
  const float* Wih1 = (const float*)d_in[5];
  const float* Whh1 = (const float*)d_in[6];
  const float* bih1 = (const float*)d_in[7];
  const float* bhh1 = (const float*)d_in[8];
  const float* W1   = (const float*)d_in[9];
  const float* b1   = (const float*)d_in[10];
  const float* W2   = (const float*)d_in[11];
  const float* b2   = (const float*)d_in[12];
  float* out = (float*)d_out;
  (void)in_sizes; (void)n_in; (void)out_size; (void)ws_size;

  // workspace layout (bytes): 256MB total
  char* ws = (char*)d_ws;
  bf16* Xbf   = (bf16*)(ws);                          //  32MB (natural (seq,batch) rows)
  bf16* Wih0b = (bf16*)(ws + ((size_t)32  << 20));    //   8MB
  bf16* Whh0b = (bf16*)(ws + ((size_t)40  << 20));    //   8MB
  bf16* Wih1b = (bf16*)(ws + ((size_t)48  << 20));    //   8MB
  bf16* Whh1b = (bf16*)(ws + ((size_t)56  << 20));    //   8MB
  bf16* XG    = (bf16*)(ws + ((size_t)64  << 20));    // 128MB tiled (reused by both layers)
  bf16* H1    = (bf16*)(ws + ((size_t)192 << 20));    //  32MB row-major
  bf16* H2    = (bf16*)(ws + ((size_t)224 << 20));    //  32MB row-major

  // bf16 conversions (vectorized x4)
  cvt_f32_bf16<<<2048, 256, 0, stream>>>((const float4*)x, (v4bf*)Xbf,
                                         MROWS * LSTM_D / 4);
  cvt_f32_bf16<<<1024, 256, 0, stream>>>((const float4*)Wih0, (v4bf*)Wih0b,
                                         LSTM_G * LSTM_D / 4);
  cvt_f32_bf16<<<1024, 256, 0, stream>>>((const float4*)Whh0, (v4bf*)Whh0b,
                                         LSTM_G * LSTM_H / 4);
  cvt_f32_bf16<<<1024, 256, 0, stream>>>((const float4*)Wih1, (v4bf*)Wih1b,
                                         LSTM_G * LSTM_H / 4);
  cvt_f32_bf16<<<1024, 256, 0, stream>>>((const float4*)Whh1, (v4bf*)Whh1b,
                                         LSTM_G * LSTM_H / 4);

  dim3 ggrid(LSTM_G / 512, MROWS / 16);   // (8, 1024)

  // layer 1
  gemm_xg<<<ggrid, 256, 0, stream>>>(Xbf, Wih0b, bih0, bhh0, XG);
  lstm_rec<<<LSTM_N / 16, 512, 0, stream>>>(XG, Whh0b, H1);
  // layer 2 (H1 stored in the seq-major row layout the GEMM expects)
  gemm_xg<<<ggrid, 256, 0, stream>>>(H1, Wih1b, bih1, bhh1, XG);
  lstm_rec<<<LSTM_N / 16, 512, 0, stream>>>(XG, Whh1b, H2);
  // FC head
  fc_head<<<LSTM_S, 512, 0, stream>>>(H2, W1, b1, W2, b2, out);
}
